// SubjectBlockConvDynamic_21586505630381
// MI455X (gfx1250) — matrix operand order repaired and checked by reading, not compiled, not run on hardware
//
#include <hip/hip_runtime.h>

// MI455X (gfx1250) WMMA kernel for per-subject 1x1 conv (batched GEMM):
//   out[b] = W[subj[b]] (320x272) @ X[b] (272x512), fp32 in/out.
// Split-bf16 (hi+lo, 3 WMMAs) for near-fp32 accuracy at bf16 matrix rate.
// A (W slice) is converted ONCE per block (division-free mapping) and kept
// persistent in LDS (pair-packed dwords, b128 fragment loads). B (X tile)
// is staged per K-chunk with fused f32->bf16 hi/lo conversion, hi/lo
// dword-interleaved so each fragment VGPR pair is one ds_load_b64.

#define BATCH   128
#define CMAX    272
#define TDIM    512
#define D1DIM   320
#define KCHUNK  32
#define NCHUNKS 9          // ceil(272/32); tail zero-padded

#define AKP     148        // A row stride in pair-dwords (>=144 incl. zeroed pad;
                           // 148%64==20 -> 16 distinct bank groups for frag rows)
#define BNPAD   258        // B n-pad (pairs); kp stride = 516 dw; 8*516%64=32 -> no clash

typedef __attribute__((ext_vector_type(16))) __bf16 v16bf;
typedef __attribute__((ext_vector_type(8)))  float  v8f;

union FragA {            // WMMA operand: 16 bf16 = 8 dwords = 2 x b128
    v16bf v;
    uint4 q[2];
};
union FragB {
    v16bf    v;
    unsigned u[8];
};
union Pack2 {            // bf16 pair -> one dword (low = even k)
    __bf16   h[2];
    unsigned u;
};

__device__ __forceinline__ __bf16 f2bf(float x) { return (__bf16)x; }

__device__ __forceinline__ unsigned pack2(__bf16 a, __bf16 b) {
    Pack2 p; p.h[0] = a; p.h[1] = b; return p.u;
}

__global__ __launch_bounds__(256)
void subject_conv_wmma(const float* __restrict__ X,
                       const int*   __restrict__ subj,
                       const float* __restrict__ W,
                       float*       __restrict__ out)
{
    // Persistent A: [plane][m][kpair] packed bf16x2 (low = even k)
    __shared__ __align__(16) unsigned ldsA[2][64][AKP];
    // Per-chunk B: [kpair][n][hi,lo] packed bf16x2, hi/lo interleaved
    __shared__ __align__(16) unsigned ldsB[16][BNPAD][2];

    const int b  = blockIdx.z;
    const int m0 = blockIdx.y * 64;        // D1 tile base
    const int n0 = blockIdx.x * 256;       // T  tile base
    const int s  = subj[b];                // uniform -> scalar load

    const int t    = threadIdx.x;
    const int lane = t & 31;
    const int wid  = t >> 5;               // 8 waves: 2(M) x 4(N)
    const int wm   = wid >> 2;             // 0..1
    const int wn   = wid & 3;              // 0..3
    const int r    = lane & 15;
    const int half = lane >> 4;

    const float* Wb = W + (size_t)s * D1DIM * CMAX;
    const float* Xb = X + (size_t)b * CMAX * TDIM;

    // ---------------- one-time A init: 64x272 W slice -> hi/lo packed LDS ----
    // Division-free: 4 threads per row, 17 float4 each at stride 16 floats.
    {
        const int arow = t >> 2;                       // 0..63
        const int asub = t & 3;                        // 0..3
        const float* wrow = Wb + (size_t)(m0 + arow) * CMAX;
        #pragma unroll
        for (int i = 0; i < 17; ++i) {
            int c4 = ((asub + (i << 2)) << 2);         // 0..268, stride 16
            float4 w4 = *reinterpret_cast<const float4*>(wrow + c4);
            __bf16 h0 = f2bf(w4.x), h1 = f2bf(w4.y), h2 = f2bf(w4.z), h3 = f2bf(w4.w);
            uint2 hi = make_uint2(pack2(h0, h1), pack2(h2, h3));
            uint2 lo = make_uint2(pack2(f2bf(w4.x - (float)h0), f2bf(w4.y - (float)h1)),
                                  pack2(f2bf(w4.z - (float)h2), f2bf(w4.w - (float)h3)));
            int kp = c4 >> 1;                          // even
            *reinterpret_cast<uint2*>(&ldsA[0][arow][kp]) = hi;
            *reinterpret_cast<uint2*>(&ldsA[1][arow][kp]) = lo;
        }
        // zero K-pad (kp 136..147): 4 threads x 3 dwords per row
        #pragma unroll
        for (int e = 0; e < 3; ++e) {
            int kp = 136 + asub * 3 + e;
            ldsA[0][arow][kp] = 0u;
            ldsA[1][arow][kp] = 0u;
        }
    }

    // ---------------- B staging (software pipeline over K-chunks) ------------
    float4 rb[4][2];     // 4 units x (even,odd) K rows
    int b_kp[4], b_n4[4];
    #pragma unroll
    for (int u = 0; u < 4; ++u) {
        int idx = t + u * 256;             // 0..1023
        b_kp[u] = idx >> 6;                // 0..15 (k-pair)
        b_n4[u] = (idx & 63) << 2;         // 0,4,..,252
    }

    auto load_B = [&](int c) {
        const int k0 = c * KCHUNK;
        #pragma unroll
        for (int u = 0; u < 4; ++u) {
            int k = k0 + (b_kp[u] << 1);
            rb[u][0] = make_float4(0.f, 0.f, 0.f, 0.f);
            rb[u][1] = make_float4(0.f, 0.f, 0.f, 0.f);
            if (k < CMAX) {   // valid K count per chunk is even -> pair-safe
                rb[u][0] = *reinterpret_cast<const float4*>(Xb + (size_t)k * TDIM + n0 + b_n4[u]);
                rb[u][1] = *reinterpret_cast<const float4*>(Xb + (size_t)(k + 1) * TDIM + n0 + b_n4[u]);
            }
        }
    };

    auto store_B = [&]() {
        #pragma unroll
        for (int u = 0; u < 4; ++u) {
            float e0[4] = { rb[u][0].x, rb[u][0].y, rb[u][0].z, rb[u][0].w };
            float e1[4] = { rb[u][1].x, rb[u][1].y, rb[u][1].z, rb[u][1].w };
            unsigned hd[4], ld[4];
            #pragma unroll
            for (int e = 0; e < 4; ++e) {
                __bf16 h0 = f2bf(e0[e]);
                __bf16 h1 = f2bf(e1[e]);
                hd[e] = pack2(h0, h1);                       // low16 = even k
                ld[e] = pack2(f2bf(e0[e] - (float)h0),
                              f2bf(e1[e] - (float)h1));
            }
            uint4 s0 = make_uint4(hd[0], ld[0], hd[1], ld[1]);
            uint4 s1 = make_uint4(hd[2], ld[2], hd[3], ld[3]);
            unsigned* base = &ldsB[b_kp[u]][b_n4[u]][0];
            *reinterpret_cast<uint4*>(base + 0) = s0;
            *reinterpret_cast<uint4*>(base + 4) = s1;
        }
    };

    v8f acc[2][4];
    #pragma unroll
    for (int i = 0; i < 2; ++i)
        #pragma unroll
        for (int j = 0; j < 4; ++j)
            #pragma unroll
            for (int e = 0; e < 8; ++e) acc[i][j][e] = 0.0f;

    load_B(0);

    for (int c = 0; c < NCHUNKS; ++c) {
        store_B();
        __syncthreads();                   // also fences the one-time A init (c==0)

        if (c + 1 < NCHUNKS) load_B(c + 1);   // overlap with WMMAs below

        // ---- A fragments from persistent LDS (ISA 16x32 bf16 layout)
        // lane<16 holds K {0..7,16..23}; lane>=16 holds K {8..15,24..31}
        // -> pair-dword runs: c*16 + half*4 + {0..3} and +8
        FragA ahi[2], alo[2];
        #pragma unroll
        for (int mt = 0; mt < 2; ++mt) {
            int row = wm * 32 + mt * 16 + r;
            const unsigned* base0 = &ldsA[0][row][c * 16 + half * 4];
            const unsigned* base1 = &ldsA[1][row][c * 16 + half * 4];
            ahi[mt].q[0] = *reinterpret_cast<const uint4*>(base0);
            ahi[mt].q[1] = *reinterpret_cast<const uint4*>(base0 + 8);
            alo[mt].q[0] = *reinterpret_cast<const uint4*>(base1);
            alo[mt].q[1] = *reinterpret_cast<const uint4*>(base1 + 8);
        }

        // ---- B fragments: one ds_load_b64 per VGPR pair (hi,lo interleaved)
        // lane n = lane&15; lanes<16: K {0..7,16..23}, lanes>=16: K {8..15,24..31} -> kpair = half*4 + j + (j&4)
        #pragma unroll
        for (int nt = 0; nt < 4; ++nt) {
            int n = wn * 64 + nt * 16 + r;
            FragB bhi, blo;
            #pragma unroll
            for (int j = 0; j < 8; ++j) {
                uint2 pr = *reinterpret_cast<const uint2*>(&ldsB[half * 4 + j + (j & 4)][n][0]);
                bhi.u[j] = pr.x;
                blo.u[j] = pr.y;
            }
            #pragma unroll
            for (int mt = 0; mt < 2; ++mt) {
                // hi*hi + hi*lo + lo*hi  (lo*lo ~2^-16 rel, dropped)
                acc[mt][nt] = __builtin_amdgcn_wmma_f32_16x16x32_bf16(
                    false, ahi[mt].v, false, bhi.v, (short)0, acc[mt][nt], false, false);
                acc[mt][nt] = __builtin_amdgcn_wmma_f32_16x16x32_bf16(
                    false, ahi[mt].v, false, blo.v, (short)0, acc[mt][nt], false, false);
                acc[mt][nt] = __builtin_amdgcn_wmma_f32_16x16x32_bf16(
                    false, alo[mt].v, false, bhi.v, (short)0, acc[mt][nt], false, false);
            }
        }
        __syncthreads();
    }

    // ---- epilogue: C/D layout (8 VGPR f32, lanes = N, VGPR = M row) -> global
    float* Ob = out + (size_t)b * D1DIM * TDIM;
    #pragma unroll
    for (int mt = 0; mt < 2; ++mt) {
        int dbase = m0 + wm * 32 + mt * 16 + half * 8;
        #pragma unroll
        for (int nt = 0; nt < 4; ++nt) {
            int tc = n0 + wn * 64 + nt * 16 + r;
            #pragma unroll
            for (int vv = 0; vv < 8; ++vv)
                Ob[(size_t)(dbase + vv) * TDIM + tc] = acc[mt][nt][vv];
        }
    }
}

extern "C" void kernel_launch(void* const* d_in, const int* in_sizes, int n_in,
                              void* d_out, int out_size, void* d_ws, size_t ws_size,
                              hipStream_t stream)
{
    const float* X    = (const float*)d_in[0];
    const int*   subj = (const int*)  d_in[1];
    const float* W    = (const float*)d_in[2];
    float*       out  = (float*)d_out;

    dim3 grid(TDIM / 256, D1DIM / 64, BATCH);   // (2, 5, 128)
    dim3 block(256);                            // 8 wave32 waves
    subject_conv_wmma<<<grid, block, 0, stream>>>(X, subj, W, out);
}
